// MHSA_901943132415
// MI455X (gfx1250) — matrix operand order repaired and checked by reading, not compiled, not run on hardware
//
#include <hip/hip_runtime.h>
#include <hip/hip_bf16.h>

typedef __attribute__((ext_vector_type(16))) _Float16 v16h;
typedef __attribute__((ext_vector_type(8)))  _Float16 v8h;
typedef __attribute__((ext_vector_type(8)))  float    v8f;
typedef __attribute__((ext_vector_type(4)))  float    f4;
typedef __attribute__((ext_vector_type(4)))  int      i4;

#define WMMA16(a,b,c) __builtin_amdgcn_wmma_f32_16x16x32_f16(false,(a),false,(b),(short)0,(c),false,false)

#if __has_builtin(__builtin_amdgcn_global_load_async_to_lds_b128)
#define HAS_ASYNC 1
#else
#define HAS_ASYNC 0
#warning "CDNA5 global_load_async_to_lds builtin unavailable: using VGPR-staging fallback"
#endif

// Copy 16 bytes global -> LDS via the ASYNCcnt-tracked direct-to-LDS DMA.
static __device__ __forceinline__ void cp_b128(const _Float16* g, _Float16* l) {
#if HAS_ASYNC
  __builtin_amdgcn_global_load_async_to_lds_b128(
      (__attribute__((address_space(1))) i4*)(_Float16*)g,
      (__attribute__((address_space(3))) i4*)l, 0, 0);
#else
  *(v8h*)l = *(const v8h*)g;
#endif
}

static __device__ __forceinline__ void cp_wait() {
#if HAS_ASYNC
#if __has_builtin(__builtin_amdgcn_s_wait_asynccnt)
  __builtin_amdgcn_s_wait_asynccnt(0);
#else
  asm volatile("s_wait_asynccnt 0x0" ::: "memory");
#endif
#endif
}

// Butterfly reductions across the 16-lane halves of a wave32 (ds_swizzle SWAPX1/2/4/8).
static __device__ __forceinline__ float bfly_max16(float x) {
  x = fmaxf(x, __int_as_float(__builtin_amdgcn_ds_swizzle(__float_as_int(x), (1 << 10) | 0x1F)));
  x = fmaxf(x, __int_as_float(__builtin_amdgcn_ds_swizzle(__float_as_int(x), (2 << 10) | 0x1F)));
  x = fmaxf(x, __int_as_float(__builtin_amdgcn_ds_swizzle(__float_as_int(x), (4 << 10) | 0x1F)));
  x = fmaxf(x, __int_as_float(__builtin_amdgcn_ds_swizzle(__float_as_int(x), (8 << 10) | 0x1F)));
  return x;
}
static __device__ __forceinline__ float bfly_sum16(float x) {
  x += __int_as_float(__builtin_amdgcn_ds_swizzle(__float_as_int(x), (1 << 10) | 0x1F));
  x += __int_as_float(__builtin_amdgcn_ds_swizzle(__float_as_int(x), (2 << 10) | 0x1F));
  x += __int_as_float(__builtin_amdgcn_ds_swizzle(__float_as_int(x), (4 << 10) | 0x1F));
  x += __int_as_float(__builtin_amdgcn_ds_swizzle(__float_as_int(x), (8 << 10) | 0x1F));
  return x;
}

// A-fragment (16x32 f16): lane -> row = lane&15; elems 0..7 k=k0+(lane>>4)*8+e,
// elems 8..15 k=k0+16+(lane>>4)*8+(e-8).
static __device__ __forceinline__ v16h ld_a_frag(const _Float16* base, int stride,
                                                 int lane, int k0) {
  const _Float16* p = base + (lane & 15) * stride + k0 + ((lane >> 4) << 3);
  v8h lo = *(const v8h*)p;
  v8h hi = *(const v8h*)(p + 16);
  v16h r;
#pragma unroll
  for (int i = 0; i < 8; ++i) { r[i] = lo[i]; r[i + 8] = hi[i]; }
  return r;
}

// B-fragment (32x16 f16) from a K-contiguous (N-major) LDS tile:
// lane -> col = lane&15; elems 0..7 k=k0+(lane>>4)*8+e, elems 8..15 k=k0+16+(lane>>4)*8+(e-8).
static __device__ __forceinline__ v16h ld_b_frag(const _Float16* base, int stride,
                                                 int lane, int k0) {
  const _Float16* p = base + (lane & 15) * stride + k0 + ((lane >> 4) << 3);
  v8h lo = *(const v8h*)p;
  v8h hi = *(const v8h*)(p + 16);
  v16h r;
#pragma unroll
  for (int i = 0; i < 8; ++i) { r[i] = lo[i]; r[i + 8] = hi[i]; }
  return r;
}

// ---------------- fp32 -> f16 convert (vectorized) ----------------
__global__ __launch_bounds__(256) void mhsa_cvt_f16(const float* __restrict__ in,
                                                    _Float16* __restrict__ out) {
  size_t i = ((size_t)blockIdx.x * 256 + threadIdx.x) * 8;
  f4 a = *(const f4*)(in + i);
  f4 b = *(const f4*)(in + i + 4);
  v8h o;
#pragma unroll
  for (int e = 0; e < 4; ++e) { o[e] = (_Float16)a[e]; o[e + 4] = (_Float16)b[e]; }
  *(v8h*)(out + i) = o;
}

// ---------------- weight convert + transpose: WT[N][K] = (f16)W[K][N] ----------
__global__ __launch_bounds__(256) void mhsa_cvt_transpose(const float* __restrict__ W,
                                                          _Float16* __restrict__ WT,
                                                          int K, int N) {
  __shared__ _Float16 t[32][33];
  const int tid = threadIdx.x;
  const int k0 = blockIdx.y * 32, n0 = blockIdx.x * 32;
#pragma unroll
  for (int p = 0; p < 4; ++p) {
    int idx = tid + p * 256;
    int kk = idx >> 5, nn = idx & 31;
    t[nn][kk] = (_Float16)W[(size_t)(k0 + kk) * N + n0 + nn];  // coalesced read
  }
  __syncthreads();
#pragma unroll
  for (int p = 0; p < 4; ++p) {
    int idx = tid + p * 256;
    int nn = idx >> 5, kk = idx & 31;
    WT[(size_t)(n0 + nn) * K + k0 + kk] = t[nn][kk];           // coalesced write
  }
}

// ---------------- all-f16 GEMM: C[M,N] = A[M,K] * BT[N,K]^T + bias --------------
// 128x128 block tile, BK=64, double-buffered LDS staged with async-to-LDS copies.
// 8 waves: wave (wm=w>>1, wn=w&1) owns 32x64 => 2x4 WMMA accums, 16 WMMA/K-step.
template <bool OUT_F16>
__global__ __launch_bounds__(256) void mhsa_gemm_f16(
    const _Float16* __restrict__ A, const _Float16* __restrict__ BT,
    const float* __restrict__ bias, _Float16* __restrict__ outH,
    float* __restrict__ outF, int M, int N, int K) {
  __shared__ __align__(16) _Float16 As[2][128][64];  // 32 KB
  __shared__ __align__(16) _Float16 Bs[2][128][64];  // 32 KB

  const int tid  = threadIdx.x;
  const int lane = tid & 31;
  const int wave = tid >> 5;
  const int m0 = blockIdx.y * 128;
  const int n0 = blockIdx.x * 128;
  const int wm = (wave >> 1) * 32;
  const int wn = (wave & 1) * 64;

  const _Float16* Ab = A + (size_t)m0 * K;
  const _Float16* Bb = BT + (size_t)n0 * K;
  const int row = tid >> 3;        // 0..31
  const int seg = (tid & 7) << 3;  // 0..56 halves

  auto stage = [&](int t, int buf) {
#pragma unroll
    for (int r = 0; r < 4; ++r) {
      int rr = row + r * 32;
      cp_b128(Ab + (size_t)rr * K + t * 64 + seg, &As[buf][rr][seg]);
      cp_b128(Bb + (size_t)rr * K + t * 64 + seg, &Bs[buf][rr][seg]);
    }
  };

  v8f acc[2][4];
#pragma unroll
  for (int i = 0; i < 2; ++i)
#pragma unroll
    for (int j = 0; j < 4; ++j) acc[i][j] = (v8f){0.f,0.f,0.f,0.f,0.f,0.f,0.f,0.f};

  stage(0, 0);
  const int nt = K >> 6;
  for (int t = 0; t < nt; ++t) {
    cp_wait();
    __syncthreads();                       // tile t visible; tile t-1 readers done
    if (t + 1 < nt) stage(t + 1, (t + 1) & 1);  // overlap copy with compute
    const int buf = t & 1;

    v16h af[2][2], bf[4][2];
#pragma unroll
    for (int i = 0; i < 2; ++i)
#pragma unroll
      for (int c = 0; c < 2; ++c)
        af[i][c] = ld_a_frag(&As[buf][wm + i * 16][0], 64, lane, c * 32);
#pragma unroll
    for (int j = 0; j < 4; ++j)
#pragma unroll
      for (int c = 0; c < 2; ++c)
        bf[j][c] = ld_b_frag(&Bs[buf][wn + j * 16][0], 64, lane, c * 32);
#pragma unroll
    for (int i = 0; i < 2; ++i)
#pragma unroll
      for (int j = 0; j < 4; ++j) {
        acc[i][j] = WMMA16(af[i][0], bf[j][0], acc[i][j]);
        acc[i][j] = WMMA16(af[i][1], bf[j][1], acc[i][j]);
      }
  }

  // Epilogue: C layout row = v + 8*(lane>>4), col = lane&15 per 16x16 tile.
#pragma unroll
  for (int i = 0; i < 2; ++i) {
    int mrow0 = m0 + wm + i * 16 + ((lane >> 4) << 3);
#pragma unroll
    for (int j = 0; j < 4; ++j) {
      int nn = n0 + wn + j * 16 + (lane & 15);
      float bv = bias[nn];
#pragma unroll
      for (int v = 0; v < 8; ++v) {
        float val = acc[i][j][v] + bv;
        size_t off = (size_t)(mrow0 + v) * N + nn;
        if (OUT_F16) outH[off] = (_Float16)val;
        else         outF[off] = val;
      }
    }
  }
}

// ---------------- Flash attention (double-buffered K/V staging) ----------------
__global__ __launch_bounds__(256) void mhsa_flash_attn(
    const _Float16* __restrict__ Qp, const _Float16* __restrict__ Kp,
    const _Float16* __restrict__ Vp, _Float16* __restrict__ AO) {
  const int T = 2048, D = 1024, DH = 64;

  __shared__ __align__(16) _Float16 Qs[128][64];      // 16 KB
  __shared__ __align__(16) _Float16 Ks[2][64][64];    // 16 KB (key-major)
  __shared__ __align__(16) _Float16 Vt[2][64][64];    // 16 KB (dh-major)
  __shared__ __align__(16) _Float16 Ps[8][16][64];    // 16 KB per-wave P scratch

  const int tid  = threadIdx.x;
  const int lane = tid & 31;
  const int wave = tid >> 5;
  const int b  = blockIdx.y >> 4;
  const int h  = blockIdx.y & 15;
  const int q0 = blockIdx.x * 128;

  const _Float16* Qg = Qp + ((size_t)b * T + q0) * D + h * DH;
  const _Float16* Kb = Kp + (size_t)b * T * D + h * DH;
  const _Float16* Vb = Vp + (size_t)b * T * D + h * DH;

  // Stage K (async) and V (VGPR transpose) tiles for key block kt0 into buf.
  auto stageKV = [&](int kt0, int buf) {
    const _Float16* Kg = Kb + (size_t)kt0 * D;
    const _Float16* Vg = Vb + (size_t)kt0 * D;
    int r0 = tid >> 3, s0 = (tid & 7) << 3;
    cp_b128(Kg + (size_t)r0 * D + s0, &Ks[buf][r0][s0]);
    cp_b128(Kg + (size_t)(r0 + 32) * D + s0, &Ks[buf][r0 + 32][s0]);
    int r = tid >> 2, d0 = (tid & 3) << 4;
    v8h x0 = *(const v8h*)(Vg + (size_t)r * D + d0);
    v8h x1 = *(const v8h*)(Vg + (size_t)r * D + d0 + 8);
#pragma unroll
    for (int e = 0; e < 8; ++e) { Vt[buf][d0 + e][r] = x0[e]; Vt[buf][d0 + 8 + e][r] = x1[e]; }
  };

  // Stage the 128x64 Q tile (async) + first K/V tile, then wait once.
  {
    int r0 = tid >> 3, s0 = (tid & 7) << 3;
#pragma unroll
    for (int r = 0; r < 4; ++r)
      cp_b128(Qg + (size_t)(r0 + r * 32) * D + s0, &Qs[r0 + r * 32][s0]);
  }
  stageKV(0, 0);
  cp_wait();
  __syncthreads();

  // Resident Q fragments with the 1/sqrt(dh) scale folded in (0.125 exact in f16).
  v16h hscale;
#pragma unroll
  for (int i = 0; i < 16; ++i) hscale[i] = (_Float16)0.125f;
  v16h aq0 = ld_a_frag(&Qs[wave * 16][0], 64, lane, 0) * hscale;
  v16h aq1 = ld_a_frag(&Qs[wave * 16][0], 64, lane, 32) * hscale;

  v8f o[4];
#pragma unroll
  for (int u = 0; u < 4; ++u) o[u] = (v8f){0.f,0.f,0.f,0.f,0.f,0.f,0.f,0.f};
  float mrow[8], lrow[8];
#pragma unroll
  for (int v = 0; v < 8; ++v) { mrow[v] = -1e30f; lrow[v] = 0.f; }

  const int NT = T / 64;
  for (int it = 0; it < NT; ++it) {
    if (it > 0) {
      cp_wait();        // tile it's async K copy complete (tile it-1's already waited)
      __syncthreads();  // all waves: tile it visible, tile it-1 compute finished
    }
    if (it + 1 < NT) stageKV((it + 1) * 64, (it + 1) & 1);  // overlap with compute
    const int buf = it & 1;

    // S tile: 16 q-rows x 64 keys = 4 C tiles, K-dim DH = 2x32 (scale pre-folded)
    v8f s[4];
#pragma unroll
    for (int t = 0; t < 4; ++t) {
      v16h b0 = ld_b_frag(&Ks[buf][t * 16][0], 64, lane, 0);
      v16h b1 = ld_b_frag(&Ks[buf][t * 16][0], 64, lane, 32);
      v8f c = (v8f){0.f,0.f,0.f,0.f,0.f,0.f,0.f,0.f};
      c = WMMA16(aq0, b0, c);
      c = WMMA16(aq1, b1, c);
      s[t] = c;
    }

    float pm[8], alpha[8], ps[8];
#pragma unroll
    for (int v = 0; v < 8; ++v)
      pm[v] = bfly_max16(fmaxf(fmaxf(s[0][v], s[1][v]), fmaxf(s[2][v], s[3][v])));
#pragma unroll
    for (int v = 0; v < 8; ++v) {
      float mn = fmaxf(mrow[v], pm[v]);
      alpha[v] = __expf(mrow[v] - mn);
      mrow[v] = mn;
      ps[v] = 0.f;
    }
#pragma unroll
    for (int t = 0; t < 4; ++t)
#pragma unroll
      for (int v = 0; v < 8; ++v) {
        float p = __expf(s[t][v] - mrow[v]);
        s[t][v] = p;
        ps[v] += p;
      }
#pragma unroll
    for (int v = 0; v < 8; ++v) {
      ps[v] = bfly_sum16(ps[v]);
      lrow[v] = lrow[v] * alpha[v] + ps[v];
    }
#pragma unroll
    for (int u = 0; u < 4; ++u)
#pragma unroll
      for (int v = 0; v < 8; ++v) o[u][v] *= alpha[v];

    // Spill P (C layout) to wave-private LDS as f16, reload as A-fragments
#pragma unroll
    for (int t = 0; t < 4; ++t)
#pragma unroll
      for (int v = 0; v < 8; ++v)
        Ps[wave][v + ((lane >> 4) << 3)][t * 16 + (lane & 15)] = (_Float16)s[t][v];

    v16h ap0 = ld_a_frag(&Ps[wave][0][0], 64, lane, 0);
    v16h ap1 = ld_a_frag(&Ps[wave][0][0], 64, lane, 32);
#pragma unroll
    for (int u = 0; u < 4; ++u) {
      v16h bv0 = ld_b_frag(&Vt[buf][u * 16][0], 64, lane, 0);
      v16h bv1 = ld_b_frag(&Vt[buf][u * 16][0], 64, lane, 32);
      o[u] = WMMA16(ap0, bv0, o[u]);
      o[u] = WMMA16(ap1, bv1, o[u]);
    }
  }

  float inv[8];
#pragma unroll
  for (int v = 0; v < 8; ++v) inv[v] = 1.f / lrow[v];
  _Float16* Og = AO + ((size_t)b * T + q0 + wave * 16) * D + h * DH;
#pragma unroll
  for (int u = 0; u < 4; ++u)
#pragma unroll
    for (int v = 0; v < 8; ++v)
      Og[(size_t)(v + ((lane >> 4) << 3)) * D + u * 16 + (lane & 15)] =
          (_Float16)(o[u][v] * inv[v]);
}

extern "C" void kernel_launch(void* const* d_in, const int* in_sizes, int n_in,
                              void* d_out, int out_size, void* d_ws, size_t ws_size,
                              hipStream_t stream) {
  const int B = 4, T = 2048, D = 1024;
  const int M = B * T;  // 8192

  const float* q  = (const float*)d_in[0];
  const float* k  = (const float*)d_in[1];
  const float* v  = (const float*)d_in[2];
  const float* Wq = (const float*)d_in[3];
  const float* bq = (const float*)d_in[4];
  const float* Wk = (const float*)d_in[5];
  const float* bk = (const float*)d_in[6];
  const float* Wv = (const float*)d_in[7];
  const float* bv = (const float*)d_in[8];
  const float* Wo = (const float*)d_in[9];
  const float* bo = (const float*)d_in[10];
  float* out = (float*)d_out;

  const size_t SZ  = (size_t)M * D;  // activation elements
  const size_t WSZ = (size_t)D * D;  // weight elements
  _Float16* Qin = (_Float16*)d_ws;
  _Float16* Kin = Qin + SZ;
  _Float16* Vin = Kin + SZ;
  _Float16* WqT = Vin + SZ;
  _Float16* WkT = WqT + WSZ;
  _Float16* WvT = WkT + WSZ;
  _Float16* WoT = WvT + WSZ;
  _Float16* Qp  = WoT + WSZ;
  _Float16* Kp  = Qp + SZ;
  _Float16* Vp  = Kp + SZ;
  _Float16* AO  = Qin;  // Qin is dead after the Q projection

  const int cvtBlocks = (int)(SZ / (256 * 8));  // 4096
  mhsa_cvt_f16<<<cvtBlocks, 256, 0, stream>>>(q, Qin);
  mhsa_cvt_f16<<<cvtBlocks, 256, 0, stream>>>(k, Kin);
  mhsa_cvt_f16<<<cvtBlocks, 256, 0, stream>>>(v, Vin);

  dim3 gt(D / 32, D / 32);  // (32,32)
  mhsa_cvt_transpose<<<gt, 256, 0, stream>>>(Wq, WqT, D, D);
  mhsa_cvt_transpose<<<gt, 256, 0, stream>>>(Wk, WkT, D, D);
  mhsa_cvt_transpose<<<gt, 256, 0, stream>>>(Wv, WvT, D, D);
  mhsa_cvt_transpose<<<gt, 256, 0, stream>>>(Wo, WoT, D, D);

  dim3 gg(D / 128, M / 128);  // (8, 64)
  mhsa_gemm_f16<true><<<gg, 256, 0, stream>>>(Qin, WqT, bq, Qp, nullptr, M, D, D);
  mhsa_gemm_f16<true><<<gg, 256, 0, stream>>>(Kin, WkT, bk, Kp, nullptr, M, D, D);
  mhsa_gemm_f16<true><<<gg, 256, 0, stream>>>(Vin, WvT, bv, Vp, nullptr, M, D, D);

  dim3 ga(T / 128, B * 16);  // (16, 64)
  mhsa_flash_attn<<<ga, 256, 0, stream>>>(Qp, Kp, Vp, AO);

  mhsa_gemm_f16<false><<<gg, 256, 0, stream>>>(AO, WoT, bo, nullptr, out, M, D, D);
}
